// RNN_31499290149187
// MI455X (gfx1250) — matrix operand order repaired and checked by reading, not compiled, not run on hardware
//
#include <hip/hip_runtime.h>

#define Bsz 8192
#define Tt  79
#define Ff  64
#define Cc  256

typedef __attribute__((ext_vector_type(16))) __bf16          v16bf;
typedef __attribute__((ext_vector_type(16))) unsigned short  v16u;
typedef __attribute__((ext_vector_type(8)))  float           v8f;

// f32 -> bf16 via native fptrunc (RNE); ISel should pick v_cvt_pk_bf16_f32 on gfx1250.
__device__ __forceinline__ unsigned short f2bf(float f) {
  __bf16 h = (__bf16)f;
  return __builtin_bit_cast(unsigned short, h);
}
__device__ __forceinline__ v16bf asbf(v16u v) { return __builtin_bit_cast(v16bf, v); }

// Build the two 16x32 bf16 A-fragments of one x row-slice (K = 0..63).
// xp points at this lane's K-run base: x[row] + t*64 + (lane>>4)*8.
__device__ __forceinline__ void load_x_frags(const float* __restrict__ xp, v16u* xa) {
#pragma unroll
  for (int kt = 0; kt < 2; ++kt) {
    const float4 f0 = *(const float4*)(xp + kt * 32);
    const float4 f1 = *(const float4*)(xp + kt * 32 + 4);
    const float4 f2 = *(const float4*)(xp + kt * 32 + 16);
    const float4 f3 = *(const float4*)(xp + kt * 32 + 20);
    v16u a;
    a[0]  = f2bf(f0.x); a[1]  = f2bf(f0.y); a[2]  = f2bf(f0.z); a[3]  = f2bf(f0.w);
    a[4]  = f2bf(f1.x); a[5]  = f2bf(f1.y); a[6]  = f2bf(f1.z); a[7]  = f2bf(f1.w);
    a[8]  = f2bf(f2.x); a[9]  = f2bf(f2.y); a[10] = f2bf(f2.z); a[11] = f2bf(f2.w);
    a[12] = f2bf(f3.x); a[13] = f2bf(f3.y); a[14] = f2bf(f3.z); a[15] = f2bf(f3.w);
    xa[kt] = a;
  }
}

extern __shared__ unsigned short smem[];  // dynamic LDS: 288 KB
// layout (in 16-bit halves):
//   [0      , 65536 )  U1 B-fragments
//   [65536  , 131072)  U2 B-fragments
//   [131072 , 139264)  h1 state, 2 buffers x 4096 halves (A-fragment layout)
//   [139264 , 147456)  h2 state, 2 buffers x 4096 halves

__global__ __launch_bounds__(512)
void rnn_fused_kernel(const float* __restrict__ x,
                      const float* __restrict__ W1, const float* __restrict__ U1,
                      const float* __restrict__ b1,
                      const float* __restrict__ W2, const float* __restrict__ U2,
                      const float* __restrict__ b2,
                      float* __restrict__ out)
{
  const int tid   = threadIdx.x;
  const int lane  = tid & 31;
  const int w     = tid >> 5;       // wave id == output-channel tile (0..15)
  const int lhi   = lane >> 4;
  const int lmod  = lane & 15;
  const int bBase = blockIdx.x * 16;

  unsigned short* u1s   = smem;
  unsigned short* u2s   = smem + 65536;
  unsigned short* hbase = smem + 131072;

  // ---- prologue: stage U1/U2 (f32 -> bf16) into LDS, pre-swizzled as WMMA B-fragments ----
  // U1/U2 B frags as staged here (this code's K order): lane l holds N=(l&15), K = 32*kt + 16*(l>>4) + j, j=0..15 contiguous.
  for (int i = tid; i < 65536; i += 512) {
    int k = i >> 8, n = i & 255;                        // U is [k_in][n_out], row-major
    int wt = n >> 4, kt = k >> 5;
    int ln = (n & 15) + (((k >> 4) & 1) << 4);
    int off = ((wt * 8 + kt) * 32 + ln) * 16 + (k & 15);
    u1s[off] = f2bf(U1[i]);
    u2s[off] = f2bf(U2[i]);
  }
  // zero initial hidden state (both parities)
  for (int i = tid; i < 16384; i += 512) hbase[i] = 0;

  // ---- per-wave register-resident W1/W2 B-fragments ----
  v16bf w1f[2];
  v16bf w2f[8];
#pragma unroll
  for (int kt = 0; kt < 2; ++kt) {
    v16u t;
#pragma unroll
    for (int j = 0; j < 16; ++j) {
      int k = kt * 32 + lhi * 8 + (j & 7) + ((j >> 3) << 4);
      t[j] = f2bf(W1[k * Cc + (w * 16 + lmod)]);
    }
    w1f[kt] = asbf(t);
  }
#pragma unroll
  for (int kt = 0; kt < 8; ++kt) {
    v16u t;
#pragma unroll
    for (int j = 0; j < 16; ++j) {
      int k = kt * 32 + lhi * 16 + j;
      t[j] = f2bf(W2[k * Cc + (w * 16 + lmod)]);
    }
    w2f[kt] = asbf(t);
  }

  const int   cIdx  = w * 16 + lmod;     // this lane's output channel
  const float bias1 = b1[cIdx];
  const float bias2 = b2[cIdx];

  // per-lane constants for scattering relu'd C-layout results into A-fragment layout
  const int sub    = cIdx & 31;
  const int kt_c   = cIdx >> 5;
  const int half_c = sub & 15;                        // j (contiguous K)
  const int g_c    = sub >> 4;                        // A-frag lane group

  __syncthreads();

  const float* xrow = x + (size_t)(bBase + lmod) * Tt * Ff + lhi * 8;

  // software pipeline: x fragments for step s=0 (t = T-1)
  v16u xa[2];
  load_x_frags(xrow + (Tt - 1) * Ff, xa);

  for (int s = 0; s < Tt; ++s) {
    const int t  = Tt - 1 - s;          // reversed-time scan; write out[:,t] directly
    const int pr = s & 1, pw = pr ^ 1;
    unsigned short* h1r = hbase +        pr * 4096;
    unsigned short* h1w = hbase +        pw * 4096;
    unsigned short* h2r = hbase + 8192 + pr * 4096;
    unsigned short* h2w = hbase + 8192 + pw * 4096;

    // ================= layer 1 =================
    v8f acc1;
#pragma unroll
    for (int r = 0; r < 8; ++r) acc1[r] = bias1;

    // h1_{t-1} @ U1 first (LDS-fed, latency overlapped across the 8 WMMAs)
#pragma unroll
    for (int kt = 0; kt < 8; ++kt) {
      v16u a  = *(const v16u*)(h1r + (kt * 32 + lane) * 16);
      v16u bb = *(const v16u*)(u1s + ((w * 8 + kt) * 32 + lane) * 16);
      acc1 = __builtin_amdgcn_wmma_f32_16x16x32_bf16(false, asbf(a), false, asbf(bb),
                                                     (short)0, acc1, false, false);
    }
    // x_t @ W1 with prefetched bf16 fragments (no load/convert on critical path)
#pragma unroll
    for (int kt = 0; kt < 2; ++kt) {
      acc1 = __builtin_amdgcn_wmma_f32_16x16x32_bf16(false, asbf(xa[kt]), false, w1f[kt],
                                                     (short)0, acc1, false, false);
    }
    // relu -> h1_new (bf16, A-fragment layout)
#pragma unroll
    for (int r = 0; r < 8; ++r) {
      float v = fmaxf(acc1[r], 0.0f);
      int   M = r + 8 * lhi;
      h1w[(kt_c * 32 + (M + (g_c << 4))) * 16 + half_c] = f2bf(v);
    }
    __syncthreads();   // h1_new visible to all waves

    // prefetch next step's x fragments; global-load latency hides under layer-2 WMMAs
    v16u xnext[2];
    {
      const int tn = (t > 0) ? (t - 1) : 0;
      load_x_frags(xrow + tn * Ff, xnext);
    }

    // ================= layer 2 =================
    v8f acc2;
#pragma unroll
    for (int r = 0; r < 8; ++r) acc2[r] = bias2;

    // h1_t @ W2 (B frags in registers)
#pragma unroll
    for (int kt = 0; kt < 8; ++kt) {
      v16u a = *(const v16u*)(h1w + (kt * 32 + lane) * 16);
      acc2 = __builtin_amdgcn_wmma_f32_16x16x32_bf16(false, asbf(a), false, w2f[kt],
                                                     (short)0, acc2, false, false);
    }
    // h2_{t-1} @ U2
#pragma unroll
    for (int kt = 0; kt < 8; ++kt) {
      v16u a  = *(const v16u*)(h2r + (kt * 32 + lane) * 16);
      v16u bb = *(const v16u*)(u2s + ((w * 8 + kt) * 32 + lane) * 16);
      acc2 = __builtin_amdgcn_wmma_f32_16x16x32_bf16(false, asbf(a), false, asbf(bb),
                                                     (short)0, acc2, false, false);
    }
    // relu -> out (f32, non-temporal: pure streaming store) and h2_new (bf16 state)
#pragma unroll
    for (int r = 0; r < 8; ++r) {
      float v = fmaxf(acc2[r], 0.0f);
      int   M = r + 8 * lhi;
      __builtin_nontemporal_store(v, out + ((size_t)(bBase + M) * Tt + t) * Cc + cIdx);
      h2w[(kt_c * 32 + (M + (g_c << 4))) * 16 + half_c] = f2bf(v);
    }
    __syncthreads();   // protect h1/h2 read buffers before next step overwrites

    xa[0] = xnext[0];
    xa[1] = xnext[1];
  }
}

extern "C" void kernel_launch(void* const* d_in, const int* in_sizes, int n_in,
                              void* d_out, int out_size, void* d_ws, size_t ws_size,
                              hipStream_t stream) {
  const float* x  = (const float*)d_in[0];
  const float* W1 = (const float*)d_in[1];
  const float* U1 = (const float*)d_in[2];
  const float* b1 = (const float*)d_in[3];
  const float* W2 = (const float*)d_in[4];
  const float* U2 = (const float*)d_in[5];
  const float* b2 = (const float*)d_in[6];
  float* out = (float*)d_out;

  const size_t ldsBytes = 294912;  // 288 KB: U1(128K)+U2(128K)+h state(32K), ISA allows 320K/WG
  rnn_fused_kernel<<<dim3(Bsz / 16), dim3(512), ldsBytes, stream>>>(
      x, W1, U1, b1, W2, U2, b2, out);
}
